// BahdanauAttention_35467839930638
// MI455X (gfx1250) — matrix operand order repaired and checked by reading, not compiled, not run on hardware
//
#include <hip/hip_runtime.h>
#include <hip/hip_bf16.h>

// Problem constants (match reference)
#define B_   32
#define S_   4096
#define K_   512    // ENC == DEC == UNITS == 512
#define U_   512

typedef __attribute__((ext_vector_type(16))) __bf16 v16bf;
typedef __attribute__((ext_vector_type(2)))  __bf16 v2bf;
typedef __attribute__((ext_vector_type(8)))  float  v8f;

union FragBF16 { v16bf v; uint4 q[2]; };

// Native truncation -> packed V_CVT_PK_BF16_F32 (verified in round-2 disasm).
__device__ __forceinline__ unsigned int pack2_bf16(float a, float b) {
  v2bf v;
  v.x = (__bf16)a;
  v.y = (__bf16)b;
  return __builtin_bit_cast(unsigned int, v);
}

__device__ __forceinline__ unsigned short f32_to_bf16(float f) {
  __bf16 h = (__bf16)f;
  return __builtin_bit_cast(unsigned short, h);
}

// Native V_TANH_F32 (TRANS32 class, co-executes with the WMMA/XDL pipe) when
// the toolchain exposes it for gfx1250; OCML tanh otherwise.
__device__ __forceinline__ float fast_tanh(float x) {
#if __has_builtin(__builtin_amdgcn_tanhf)
  return __builtin_amdgcn_tanhf(x);
#else
  return tanhf(x);
#endif
}

// ---------------- Kernel 0: We [K,U] f32 row-major -> WeT [U,K] bf16 ----------------
__global__ __launch_bounds__(256)
void prep_weT(const float* __restrict__ We, unsigned short* __restrict__ WeT) {
  int idx = blockIdx.x * 256 + threadIdx.x;   // K_*U_ threads
  int k = idx >> 9;
  int n = idx & 511;
  WeT[(size_t)n * K_ + k] = f32_to_bf16(We[(size_t)k * U_ + n]);
}

// ---------------- Kernel 1: dec = H_decoder @ Wd + bd -> [B, U] f32 ----------------
__global__ __launch_bounds__(256)
void dec_proj(const float* __restrict__ Hd, const float* __restrict__ Wd,
              const float* __restrict__ bd, float* __restrict__ dec) {
  int idx = blockIdx.x * 256 + threadIdx.x;   // B_*U_ threads
  int b = idx >> 9;
  int u = idx & 511;
  const float* h = Hd + (size_t)b * K_;
  float acc = bd[u];
  #pragma unroll 8
  for (int k = 0; k < K_; ++k) acc = fmaf(h[k], Wd[(size_t)k * U_ + u], acc);
  dec[idx] = acc;
}

// ---------------- Kernel 2: WMMA GEMM + tanh + Wc-dot -> scores [B, S] ----------------
// Block: 256 threads (8 wave32). Tile: M=64 rows of H_encoder, N=512 units
// (each wave owns 64 columns). K loop in steps of 32 (bf16 WMMA K).
__global__ __launch_bounds__(256)
void score_kernel(const float* __restrict__ Henc,
                  const unsigned short* __restrict__ WeT,
                  const float* __restrict__ dec,
                  const float* __restrict__ be,
                  const float* __restrict__ Wc,
                  const float* __restrict__ bc,
                  float* __restrict__ scores) {
  __shared__ unsigned short sA[64 * 32];   // 64 rows x 32 K, bf16 (4 KB)
  __shared__ float s_score[64];

  const int b      = blockIdx.y;
  const int s0     = blockIdx.x * 64;
  const int t      = threadIdx.x;
  const int lane   = t & 31;
  const int wave   = t >> 5;          // 0..7
  const int n0     = wave * 64;       // this wave's first unit column
  const int lanelo = lane & 15;
  const int lanehi = lane >> 4;       // 0 or 1

  if (t < 64) s_score[t] = 0.0f;

  const float* Ablk = Henc + ((size_t)b * S_ + s0) * (size_t)K_;
  const int arow = t >> 2;            // 0..63  (row this thread stages)
  const int acol = (t & 3) * 8;       // 0,8,16,24 (8 f32 per thread)

  v8f acc[4][4];
  #pragma unroll
  for (int mt = 0; mt < 4; ++mt)
    #pragma unroll
    for (int nt = 0; nt < 4; ++nt) {
      v8f z = {};
      acc[mt][nt] = z;
    }

  for (int k0 = 0; k0 < K_; k0 += 32) {
    // ---- stage A tile (64 x 32 f32 -> bf16) into LDS ----
    const float* src = Ablk + (size_t)arow * K_ + (k0 + acol);
    float4 x0 = *(const float4*)(src);
    float4 x1 = *(const float4*)(src + 4);
    if (k0 + 32 < K_) __builtin_prefetch((const void*)(src + 32), 0, 1);
    uint4 pk;
    pk.x = pack2_bf16(x0.x, x0.y);
    pk.y = pack2_bf16(x0.z, x0.w);
    pk.z = pack2_bf16(x1.x, x1.y);
    pk.w = pack2_bf16(x1.z, x1.w);
    *(uint4*)&sA[arow * 32 + acol] = pk;
    __syncthreads();

    // ---- B fragments straight from L2-resident WeT (32x16 bf16 layout:
    //      lanes 0-15: col=lane, K=k0..k0+15 ; lanes 16-31: col=lane-16, K=k0+16..k0+31)
    FragBF16 bf[4];
    const int kb = k0 + (lanehi << 4);
    #pragma unroll
    for (int nt = 0; nt < 4; ++nt) {
      const unsigned short* p = WeT + (size_t)(n0 + nt * 16 + lanelo) * K_ + kb;
      bf[nt].q[0] = *(const uint4*)(p);
      bf[nt].q[1] = *(const uint4*)(p + 8);
    }

    // ---- A fragments from LDS (16x32 bf16 layout:
    //      lanes 0-15: row=lane,   K=0..15
    //      lanes16-31: row=lane-16,K=16..31)
    #pragma unroll
    for (int mt = 0; mt < 4; ++mt) {
      FragBF16 af;
      const unsigned short* ap = &sA[(mt * 16 + lanelo) * 32 + (lanehi << 4)];
      af.q[0] = *(const uint4*)(ap);
      af.q[1] = *(const uint4*)(ap + 8);
      #pragma unroll
      for (int nt = 0; nt < 4; ++nt) {
        acc[mt][nt] = __builtin_amdgcn_wmma_f32_16x16x32_bf16(
            false, af.v, false, bf[nt].v, (short)0, acc[mt][nt], false, false);
      }
    }
    __syncthreads();
  }

  // ---- epilogue: score_row += sum_n tanh(acc + dec[b,n] + be[n]) * Wc[n] ----
  float addv[4], wcv[4];
  #pragma unroll
  for (int nt = 0; nt < 4; ++nt) {
    int n = n0 + nt * 16 + lanelo;
    addv[nt] = dec[(size_t)b * U_ + n] + be[n];
    wcv[nt]  = Wc[n];
  }
  // C layout: lanes 0-15 -> M = v, N = lane ; lanes 16-31 -> M = v+8, N = lane-16
  #pragma unroll
  for (int mt = 0; mt < 4; ++mt) {
    #pragma unroll
    for (int v = 0; v < 8; ++v) {
      float p = 0.0f;
      #pragma unroll
      for (int nt = 0; nt < 4; ++nt)
        p += fast_tanh(acc[mt][nt][v] + addv[nt]) * wcv[nt];
      // reduce across the 16 lanes that share this row
      p += __shfl_xor(p, 8, 32);
      p += __shfl_xor(p, 4, 32);
      p += __shfl_xor(p, 2, 32);
      p += __shfl_xor(p, 1, 32);
      if (lanelo == 0)
        atomicAdd(&s_score[mt * 16 + v + (lanehi << 3)], p);   // ds_add_f32
    }
  }
  __syncthreads();
  if (t < 64) scores[(size_t)b * S_ + s0 + t] = s_score[t] + bc[0];
}

// ---------------- Kernel 3: softmax over S per batch; emit weights + context ----------------
__global__ __launch_bounds__(256)
void softmax_out(const float* __restrict__ scores, const float* __restrict__ dec,
                 float* __restrict__ out_ctx, float* __restrict__ out_w) {
  __shared__ float red[256];
  const int b = blockIdx.x;
  const int t = threadIdx.x;
  const float* sc = scores + (size_t)b * S_;

  float m = -3.402823466e+38f;
  for (int i = t; i < S_; i += 256) m = fmaxf(m, sc[i]);
  red[t] = m; __syncthreads();
  for (int off = 128; off > 0; off >>= 1) {
    if (t < off) red[t] = fmaxf(red[t], red[t + off]);
    __syncthreads();
  }
  const float mx = red[0];
  __syncthreads();

  float s = 0.0f;
  for (int i = t; i < S_; i += 256) s += __expf(sc[i] - mx);
  red[t] = s; __syncthreads();
  for (int off = 128; off > 0; off >>= 1) {
    if (t < off) red[t] += red[t + off];
    __syncthreads();
  }
  const float inv = 1.0f / red[0];
  __syncthreads();

  float wsum = 0.0f;
  for (int i = t; i < S_; i += 256) {
    float w = __expf(sc[i] - mx) * inv;
    out_w[(size_t)b * S_ + i] = w;
    wsum += w;
  }
  red[t] = wsum; __syncthreads();
  for (int off = 128; off > 0; off >>= 1) {
    if (t < off) red[t] += red[t + off];
    __syncthreads();
  }
  const float tot = red[0];  // ~= 1.0 ; faithful to reference's context = dec * sum(w)
  for (int u = t; u < U_; u += 256)
    out_ctx[(size_t)b * U_ + u] = dec[(size_t)b * U_ + u] * tot;
}

extern "C" void kernel_launch(void* const* d_in, const int* in_sizes, int n_in,
                              void* d_out, int out_size, void* d_ws, size_t ws_size,
                              hipStream_t stream) {
  const float* Henc = (const float*)d_in[0];  // [B, S, ENC]
  const float* Hdec = (const float*)d_in[1];  // [B, DEC]
  const float* We   = (const float*)d_in[2];  // [ENC, U]
  const float* be   = (const float*)d_in[3];  // [U]
  const float* Wd   = (const float*)d_in[4];  // [DEC, U]
  const float* bd   = (const float*)d_in[5];  // [U]
  const float* Wc   = (const float*)d_in[6];  // [U, 1]
  const float* bc   = (const float*)d_in[7];  // [1]

  char* ws = (char*)d_ws;
  unsigned short* WeT = (unsigned short*)ws;                       // 512 KB bf16 [U][K]
  float* dec   = (float*)(ws + 512 * 1024);                        //  64 KB f32  [B][U]
  float* score = (float*)(ws + 512 * 1024 + 64 * 1024);            // 512 KB f32  [B][S]

  float* out_ctx = (float*)d_out;        // [B, U]   (first output)
  float* out_w   = out_ctx + B_ * U_;    // [B, S, 1] (second output)

  prep_weT  <<<dim3((K_ * U_) / 256), dim3(256), 0, stream>>>(We, WeT);
  dec_proj  <<<dim3((B_ * U_) / 256), dim3(256), 0, stream>>>(Hdec, Wd, bd, dec);
  score_kernel<<<dim3(S_ / 64, B_), dim3(256), 0, stream>>>(Henc, WeT, dec, be, Wc, bc, score);
  softmax_out <<<dim3(B_), dim3(256), 0, stream>>>(score, dec, out_ctx, out_w);
}
